// EdgeProposer_63058709840240
// MI455X (gfx1250) — compile-verified, matrix operand order checked by reading, not run on hardware
//
#include <hip/hip_runtime.h>
#include <hip/hip_bf16.h>
#include <math.h>

typedef __attribute__((ext_vector_type(16))) __bf16 v16bf;
typedef __attribute__((ext_vector_type(8)))  __bf16 v8bf;
typedef __attribute__((ext_vector_type(8)))  float  v8f;
typedef __attribute__((ext_vector_type(4)))  unsigned int v4u;

#define DIM   512
#define TD    1536            // 3*DIM
#define NCOL1 320             // 128 (prop) + 64 (dir) + 128 (rel)
#define KT1   48              // 1536 / 32
#define CT1   20              // 320 / 16
#define STRA  1544            // LDS A row stride (bf16): 16B aligned, ≡4 mod 64 dwords -> conflict-free
#define STRH  328             // LDS H row stride (bf16)
#define MROWS 32              // rows per block (2 row tiles of 16)

// dynamic LDS partition (bytes)
#define A_BYTES  (MROWS * STRA * 2)          // 98816
#define H_OFF    A_BYTES
#define H_BYTES  (MROWS * STRH * 2)          // 20992
#define H2_OFF   (H_OFF + H_BYTES)           // 119808
#define H2_BYTES (MROWS * 65 * 4)            // 8320
#define SMEM_TOTAL (H2_OFF + H2_BYTES)       // 128128

union V16U { v16bf v; struct { v8bf lo; v8bf hi; } s; };
union BC8  { v8bf b; v4u u; };

__device__ __forceinline__ v8bf cvt8(float4 a, float4 b) {
    v8bf r;
    r[0] = (__bf16)a.x; r[1] = (__bf16)a.y; r[2] = (__bf16)a.z; r[3] = (__bf16)a.w;
    r[4] = (__bf16)b.x; r[5] = (__bf16)b.y; r[6] = (__bf16)b.z; r[7] = (__bf16)b.w;
    return r;
}

// clear bf16 sign bits (abs) with integer ANDs -- avoids f32 fabs isel pattern
__device__ __forceinline__ v8bf absmask8(v8bf x) {
    BC8 c; c.b = x;
    c.u = c.u & 0x7fff7fffu;
    return c.b;
}

// ---- pack fused first-layer weights [1536 x 320] into WMMA-B bf16 fragments ----
__global__ void pack_w1(const float* __restrict__ pw1, const float* __restrict__ dw1,
                        const float* __restrict__ rw1, __bf16* __restrict__ dst) {
    int i = blockIdx.x * 256 + threadIdx.x;           // over 1536*320
    if (i >= TD * NCOL1) return;
    int k = i / NCOL1, n = i % NCOL1;
    float v;
    if (n < 128)      v = pw1[k * 128 + n];
    else if (n < 192) v = dw1[k * 64 + (n - 128)];
    else              v = rw1[k * 128 + (n - 192)];
    int kt = k >> 5, kk = k & 31;
    int ct = n >> 4, nn = n & 15;
    int lane = nn + 16 * ((kk >> 3) & 1);             // K 8..15 / 24..31 live in lanes 16..31
    int pos  = (kk >= 16 ? 8 : 0) + (kk & 7);         // K 16..31 -> vector elements 8..15
    dst[(((kt * CT1 + ct) * 32) + lane) * 16 + pos] = (__bf16)v;
}

// ---- pack second-layer weights pw2 / rw2 ([128 x 64] each) ----
__global__ void pack_w2(const float* __restrict__ pw2, const float* __restrict__ rw2,
                        __bf16* __restrict__ dst) {
    int i = blockIdx.x * 256 + threadIdx.x;           // over 2*128*64
    if (i >= 2 * 128 * 64) return;
    int which = i / (128 * 64);
    int rem   = i % (128 * 64);
    int k = rem / 64, n = rem % 64;
    float v = which ? rw2[k * 64 + n] : pw2[k * 64 + n];
    int kt = k >> 5, kk = k & 31;
    int ct = n >> 4, nn = n & 15;
    int lane = nn + 16 * ((kk >> 3) & 1);
    int pos  = (kk >= 16 ? 8 : 0) + (kk & 7);
    dst[((((which * 4 + kt) * 4 + ct) * 32) + lane) * 16 + pos] = (__bf16)v;
}

__global__ __launch_bounds__(128)
void edge_kernel(const float* __restrict__ beliefs,
                 const int* __restrict__ src_idx, const int* __restrict__ tgt_idx,
                 const float* __restrict__ pb1, const float* __restrict__ db1,
                 const float* __restrict__ rb1,
                 const float* __restrict__ pb2, const float* __restrict__ pw3,
                 const float* __restrict__ pb3,
                 const float* __restrict__ dw2, const float* __restrict__ db2,
                 const float* __restrict__ rb2,
                 const __bf16* __restrict__ B1, const __bf16* __restrict__ B2,
                 float* __restrict__ out, int N)
{
    extern __shared__ char smem[];
    __bf16* ldsA  = (__bf16*)smem;                    // 32 x 1536 bf16 feature tile (padded)
    __bf16* ldsH  = (__bf16*)(smem + H_OFF);          // 32 x 320 relu'd hidden (padded)
    float*  ldsH2 = (float*)(smem + H2_OFF);          // 32 x 64 proposal hidden2

    const int lane     = threadIdx.x & 31;
    const int wave     = threadIdx.x >> 5;
    const int lane16   = lane & 15;
    const int laneHalf = lane >> 4;
    const int rowBase  = blockIdx.x * MROWS;

    // ---------- build bf16 feature tile in LDS: [src | tgt | abs(src-tgt)] ----------
    for (int rr = 0; rr < 8; ++rr) {
        const int r = wave * 8 + rr;
        const int n = rowBase + r;
        const float* srow = beliefs + (long)src_idx[n] * DIM;
        const float* trow = beliefs + (long)tgt_idx[n] * DIM;
        const int d0 = lane * 16;                       // 16 floats per lane per row
        __bf16* arow = ldsA + r * STRA;
        const float4* s4 = (const float4*)(srow + d0);
        const float4* t4 = (const float4*)(trow + d0);
        float4 s0 = s4[0], s1 = s4[1], s2 = s4[2], s3 = s4[3];
        float4 t0 = t4[0], t1 = t4[1], t2 = t4[2], t3 = t4[3];
        *(v8bf*)(arow + d0)           = cvt8(s0, s1);
        *(v8bf*)(arow + d0 + 8)       = cvt8(s2, s3);
        *(v8bf*)(arow + DIM + d0)     = cvt8(t0, t1);
        *(v8bf*)(arow + DIM + d0 + 8) = cvt8(t2, t3);
        float4 d0v = make_float4(s0.x - t0.x, s0.y - t0.y, s0.z - t0.z, s0.w - t0.w);
        float4 d1v = make_float4(s1.x - t1.x, s1.y - t1.y, s1.z - t1.z, s1.w - t1.w);
        float4 d2v = make_float4(s2.x - t2.x, s2.y - t2.y, s2.z - t2.z, s2.w - t2.w);
        float4 d3v = make_float4(s3.x - t3.x, s3.y - t3.y, s3.z - t3.z, s3.w - t3.w);
        *(v8bf*)(arow + 2*DIM + d0)     = absmask8(cvt8(d0v, d1v));
        *(v8bf*)(arow + 2*DIM + d0 + 8) = absmask8(cvt8(d2v, d3v));
    }
    __syncthreads();

    // ---------- stage 1: feats[32x1536] @ W1[1536x320] ----------
    // each wave: 2 row tiles x 5 col tiles; B fragment loaded once, used twice
    v8f acc[2][5];
    #pragma unroll
    for (int rt = 0; rt < 2; ++rt)
        #pragma unroll
        for (int c = 0; c < 5; ++c) acc[rt][c] = (v8f){};

    const int ct0 = wave * 5;
    for (int kt = 0; kt < KT1; ++kt) {
        const __bf16* ap = ldsA + lane16 * STRA + kt * 32 + laneHalf * 8;
        V16U a0, a1;
        a0.s.lo = *(const v8bf*)(ap);                   // rows 0..15
        a0.s.hi = *(const v8bf*)(ap + 16);
        a1.s.lo = *(const v8bf*)(ap + 16 * STRA);       // rows 16..31
        a1.s.hi = *(const v8bf*)(ap + 16 * STRA + 16);
        #pragma unroll
        for (int c = 0; c < 5; ++c) {
            const __bf16* bp = B1 + (((kt * CT1 + ct0 + c) * 32) + lane) * 16;
            V16U b;
            b.s.lo = *(const v8bf*)(bp);
            b.s.hi = *(const v8bf*)(bp + 8);
            acc[0][c] = __builtin_amdgcn_wmma_f32_16x16x32_bf16(
                            false, a0.v, false, b.v, (short)0, acc[0][c], false, false);
            acc[1][c] = __builtin_amdgcn_wmma_f32_16x16x32_bf16(
                            false, a1.v, false, b.v, (short)0, acc[1][c], false, false);
        }
    }

    // bias + relu -> bf16 hidden tile in LDS (both row tiles)
    #pragma unroll
    for (int c = 0; c < 5; ++c) {
        const int colBase = (ct0 + c) * 16;
        const float* bsrc; int cb;
        if (colBase < 128)      { bsrc = pb1; cb = colBase; }
        else if (colBase < 192) { bsrc = db1; cb = colBase - 128; }
        else                    { bsrc = rb1; cb = colBase - 192; }
        const float bias = bsrc[cb + lane16];
        const int col = colBase + lane16;
        #pragma unroll
        for (int rt = 0; rt < 2; ++rt) {
            #pragma unroll
            for (int vr = 0; vr < 8; ++vr) {
                const int row = rt * 16 + vr + 8 * laneHalf;
                float v = acc[rt][c][vr] + bias;
                ldsH[row * STRH + col] = (__bf16)(v > 0.0f ? v : 0.0f);
            }
        }
    }
    __syncthreads();

    // ---------- stage 2: waves 0-1: h_p@pw2 ; waves 2-3: r@rw2 (K=128, 2 row tiles) ----------
    const int which   = wave >> 1;              // 0 = proposal, 1 = relation
    const int baseCol = which ? 192 : 0;
    const int ct2     = (wave & 1) * 2;
    v8f acc2[2][2];
    #pragma unroll
    for (int rt = 0; rt < 2; ++rt) { acc2[rt][0] = (v8f){}; acc2[rt][1] = (v8f){}; }

    #pragma unroll
    for (int kt = 0; kt < 4; ++kt) {
        const __bf16* ap = ldsH + lane16 * STRH + baseCol + kt * 32 + laneHalf * 8;
        V16U a0, a1;
        a0.s.lo = *(const v8bf*)(ap);
        a0.s.hi = *(const v8bf*)(ap + 16);
        a1.s.lo = *(const v8bf*)(ap + 16 * STRH);
        a1.s.hi = *(const v8bf*)(ap + 16 * STRH + 16);
        #pragma unroll
        for (int c = 0; c < 2; ++c) {
            const __bf16* bp = B2 + ((((which * 4 + kt) * 4 + (ct2 + c)) * 32) + lane) * 16;
            V16U b;
            b.s.lo = *(const v8bf*)(bp);
            b.s.hi = *(const v8bf*)(bp + 8);
            acc2[0][c] = __builtin_amdgcn_wmma_f32_16x16x32_bf16(
                             false, a0.v, false, b.v, (short)0, acc2[0][c], false, false);
            acc2[1][c] = __builtin_amdgcn_wmma_f32_16x16x32_bf16(
                             false, a1.v, false, b.v, (short)0, acc2[1][c], false, false);
        }
    }

    if (which == 0) {
        // proposal hidden2: relu(acc + pb2) -> LDS
        #pragma unroll
        for (int c = 0; c < 2; ++c) {
            const int col = (ct2 + c) * 16 + lane16;
            const float bias = pb2[col];
            #pragma unroll
            for (int rt = 0; rt < 2; ++rt) {
                #pragma unroll
                for (int vr = 0; vr < 8; ++vr) {
                    const int row = rt * 16 + vr + 8 * laneHalf;
                    float v = acc2[rt][c][vr] + bias;
                    ldsH2[row * 65 + col] = v > 0.0f ? v : 0.0f;
                }
            }
        }
    } else {
        // relations: acc + rb2 -> out[2N + n*64 + col]
        float* rel = out + 2L * N + (long)rowBase * 64;
        #pragma unroll
        for (int c = 0; c < 2; ++c) {
            const int col = (ct2 + c) * 16 + lane16;
            const float bias = rb2[col];
            #pragma unroll
            for (int rt = 0; rt < 2; ++rt) {
                #pragma unroll
                for (int vr = 0; vr < 8; ++vr) {
                    const int row = rt * 16 + vr + 8 * laneHalf;
                    rel[(long)row * 64 + col] = acc2[rt][c][vr] + bias;
                }
            }
        }
    }
    __syncthreads();

    // ---------- tail heads: 64-wide dots + sigmoid (one row per lane) ----------
    if (wave == 0) {
        const int row = lane;                           // 32 rows, 32 lanes
        float s = 0.0f;
        for (int c = 0; c < 64; ++c) s += ldsH2[row * 65 + c] * pw3[c];
        const float x = s + pb3[0];
        out[rowBase + row] = 1.0f / (1.0f + __expf(-x));
    }
    if (wave == 1) {
        const int row = lane;
        float s = 0.0f;
        for (int c = 0; c < 64; ++c) s += (float)ldsH[row * STRH + 128 + c] * dw2[c];
        const float x = s + db2[0];
        out[N + rowBase + row] = (1.0f / (1.0f + __expf(-x))) * 1.57079632679489662f;
    }
}

extern "C" void kernel_launch(void* const* d_in, const int* in_sizes, int n_in,
                              void* d_out, int out_size, void* d_ws, size_t ws_size,
                              hipStream_t stream) {
    // setup_inputs order:
    // 0 beliefs, 1 src_idx, 2 tgt_idx, 3 pw1, 4 pb1, 5 pw2, 6 pb2, 7 pw3, 8 pb3,
    // 9 dw1, 10 db1, 11 dw2, 12 db2, 13 rw1, 14 rb1, 15 rw2, 16 rb2
    const float* beliefs = (const float*)d_in[0];
    const int*   src_idx = (const int*)d_in[1];
    const int*   tgt_idx = (const int*)d_in[2];
    const float* pw1 = (const float*)d_in[3];
    const float* pb1 = (const float*)d_in[4];
    const float* pw2 = (const float*)d_in[5];
    const float* pb2 = (const float*)d_in[6];
    const float* pw3 = (const float*)d_in[7];
    const float* pb3 = (const float*)d_in[8];
    const float* dw1 = (const float*)d_in[9];
    const float* db1 = (const float*)d_in[10];
    const float* dw2 = (const float*)d_in[11];
    const float* db2 = (const float*)d_in[12];
    const float* rw1 = (const float*)d_in[13];
    const float* rb1 = (const float*)d_in[14];
    const float* rw2 = (const float*)d_in[15];
    const float* rb2 = (const float*)d_in[16];

    const int N = in_sizes[1];

    __bf16* B1 = (__bf16*)d_ws;                                   // 1536*320 bf16 = 983040 B
    __bf16* B2 = (__bf16*)((char*)d_ws + (size_t)TD * NCOL1 * 2); // 2*128*64 bf16 = 32768 B

    pack_w1<<<(TD * NCOL1 + 255) / 256, 256, 0, stream>>>(pw1, dw1, rw1, B1);
    pack_w2<<<(2 * 128 * 64 + 255) / 256, 256, 0, stream>>>(pw2, rw2, B2);

    edge_kernel<<<N / MROWS, 128, SMEM_TOTAL, stream>>>(
        beliefs, src_idx, tgt_idx,
        pb1, db1, rb1, pb2, pw3, pb3, dw2, db2, rb2,
        B1, B2, (float*)d_out, N);
}
